// Generator_74929999446272
// MI455X (gfx1250) — matrix operand order repaired and checked by reading, not compiled, not run on hardware
//
#include <hip/hip_runtime.h>
#include <hip/hip_bf16.h>

typedef __attribute__((ext_vector_type(16))) _Float16 v16h;
typedef __attribute__((ext_vector_type(8)))  float    v8f;

#define N_SAMP   32768
#define N_BANDS  128
#define CH       128
#define ENC      512
#define KSP      128
#define KERNEL_L 1024
#define LATENT   128
#define FB_K     512

// ---------------------------------------------------------------------------
// Stage A: fixed filter bank conv (pad 255/256), |.|, pool 256 -> pooled[128,128]
// ---------------------------------------------------------------------------
__global__ void fb_pool_kernel(const float* __restrict__ x, const float* __restrict__ fb,
                               float* __restrict__ pooled) {
  int band = blockIdx.x, frame = blockIdx.y;
  int s = threadIdx.x;                       // 0..255 -> sample within frame
  int t = frame * 256 + s;
  const float* w = fb + band * FB_K;
  float acc = 0.f;
  for (int k = 0; k < FB_K; ++k) {
    int xi = t + k - (FB_K / 2 - 1);
    float xv = (xi >= 0 && xi < N_SAMP) ? x[xi] : 0.f;
    acc += xv * w[k];
  }
  acc = fabsf(acc);
  __shared__ float red[256];
  red[s] = acc; __syncthreads();
  for (int o = 128; o > 0; o >>= 1) { if (s < o) red[s] += red[s + o]; __syncthreads(); }
  if (s == 0) pooled[band * 128 + frame] = red[0] * (1.f / 256.f);
}

// ---------------------------------------------------------------------------
// Pack f32 weights (Cout,128,taps) into f16 wave32 A-fragments:
//   wp[(((mtile*chunks)+chunk)*32 + lane)*16 + j]
// with the ISA A(16x32 f16) layout: g=lane>>4, m=mtile*16+(lane&15),
//   K = ((vg<4)?2vg:8+2vg) + (j&1) + 8g, kk = chunk*32+K, cin=kk&127, tap=kk>>7
// One 32B vector load per wave then fetches a whole A fragment, coalesced.
// ---------------------------------------------------------------------------
__global__ void pack_w_kernel(const float* __restrict__ w, _Float16* __restrict__ wp,
                              int Cout, int taps) {
  int i = blockIdx.x * blockDim.x + threadIdx.x;
  int chunks = (128 * taps) >> 5;
  int total = (Cout >> 4) * chunks * 32 * 16;
  if (i >= total) return;
  int j = i & 15;
  int lane = (i >> 4) & 31;
  int mc = i >> 9;
  int chunk = mc % chunks;
  int mtile = mc / chunks;
  int g = lane >> 4;
  int vg = j >> 1;
  int ka = ((vg < 4) ? 2 * vg : 8 + 2 * vg) + (j & 1) + (g ? 8 : 0);
  int kk = chunk * 32 + ka;
  int cin = kk & 127, tap = kk >> 7;
  int m = mtile * 16 + (lane & 15);
  wp[i] = (_Float16)w[((size_t)m * 128 + cin) * taps + tap];
}

// ---------------------------------------------------------------------------
// Generic Cin=128 conv as WMMA GEMM.  out[m,t] = b[m] + sum_{cin,tap} w[m,cin,tap] *
// in[cin, src(t + (tap-pad)*dil)], optional x2 nearest-upsample of input.
// One wave32 computes a 16(out-ch) x 64(time) strip: 4 accumulators reuse each
// A fragment (1 coalesced 32B load) across 4 v_wmma_f32_16x16x32_f16 issues.
// B fragment layout: n = lane&15, K = ((j<8)?j:j+8) + 8*(lane>>4); B loads are coalesced
// (16 consecutive time positions across lanes 0-15 of each half-wave).
// ---------------------------------------------------------------------------
__global__ __launch_bounds__(32)
void conv_wmma_kernel(const float* __restrict__ in, const _Float16* __restrict__ wp,
                      const float* __restrict__ bias, float* __restrict__ out,
                      int Cout, int Lout, int Lin, int taps, int dil,
                      int upsample, int fuseLrelu) {
  const int Cin = 128;
  in  += (size_t)blockIdx.z * Cin * Lin;     // atom batch
  out += (size_t)blockIdx.z * Cout * Lout;
  int lane = threadIdx.x;
  int nt0  = blockIdx.x * 64;                // 4 time tiles per wave
  int m0   = blockIdx.y * 16;
  int g  = lane >> 4;
  int nB = lane & 15;
  int pad = (taps - 1) >> 1;
  int Lv  = upsample ? (Lin << 1) : Lin;     // virtual (post-upsample) input length
  int chunks = (Cin * taps) >> 5;
  __builtin_prefetch(wp + ((size_t)(m0 >> 4) * chunks * 32) * 16, 0, 0);  // global_prefetch_b8
  v8f acc[4] = {};
  for (int chunk = 0; chunk < chunks; ++chunk) {
    v16h a = *(const v16h*)(wp + (((size_t)(m0 >> 4) * chunks + chunk) * 32 + lane) * 16);
#pragma unroll
    for (int nt = 0; nt < 4; ++nt) {
      int t0 = nt0 + nt * 16;
      v16h b;
#pragma unroll
      for (int j = 0; j < 16; ++j) {
        int kb = chunk * 32 + ((j < 8) ? j : j + 8) + (g ? 8 : 0);
        int cinB = kb & 127, tapB = kb >> 7;
        int t = t0 + nB + (tapB - pad) * dil;
        float bv = 0.f;
        if (t >= 0 && t < Lv) bv = in[(size_t)cinB * Lin + (upsample ? (t >> 1) : t)];
        b[j] = (_Float16)bv;
      }
      acc[nt] = __builtin_amdgcn_wmma_f32_16x16x32_f16(false, a, false, b, (short)0,
                                                       acc[nt], false, false);
    }
  }
#pragma unroll
  for (int nt = 0; nt < 4; ++nt) {
#pragma unroll
    for (int v = 0; v < 8; ++v) {
      int m = m0 + v + (g ? 8 : 0);
      int n = nt0 + nt * 16 + nB;
      if (n < Lout) {
        float r = acc[nt][v] + bias[m];
        if (fuseLrelu) r = r > 0.f ? r : 0.2f * r;
        out[(size_t)m * Lout + n] = r;
      }
    }
  }
}

// ---------------------------------------------------------------------------
// elementwise / norm helpers
// ---------------------------------------------------------------------------
__global__ void add_lrelu_kernel(const float* __restrict__ a, const float* __restrict__ resid,
                                 float* __restrict__ out, int n) {
  int i = blockIdx.x * blockDim.x + threadIdx.x;
  if (i >= n) return;
  float v = a[i] + resid[i];
  out[i] = v > 0.f ? v : 0.2f * v;
}

__global__ void bn_stats_kernel(const float* __restrict__ x, float* __restrict__ stats,
                                int C, int L, int A) {
  int c = blockIdx.x, tid = threadIdx.x;
  float s = 0.f, s2 = 0.f;
  for (int a = 0; a < A; ++a) {
    const float* p = x + ((size_t)a * C + c) * L;
    for (int t = tid; t < L; t += blockDim.x) { float v = p[t]; s += v; s2 += v * v; }
  }
  __shared__ float r1[256], r2[256];
  r1[tid] = s; r2[tid] = s2; __syncthreads();
  for (int o = 128; o > 0; o >>= 1) {
    if (tid < o) { r1[tid] += r1[tid + o]; r2[tid] += r2[tid + o]; }
    __syncthreads();
  }
  if (tid == 0) {
    float n = (float)A * (float)L;
    float m = r1[0] / n;
    stats[c] = m; stats[C + c] = r2[0] / n - m * m;
  }
}

__global__ void bn_apply_kernel(float* __restrict__ x, const float* __restrict__ stats,
                                const float* __restrict__ g, const float* __restrict__ be,
                                int C, int L, long long n, int lrelu) {
  long long i = (long long)blockIdx.x * blockDim.x + threadIdx.x;
  if (i >= n) return;
  int c = (int)((i / L) % C);
  float y = (x[i] - stats[c]) * rsqrtf(stats[C + c] + 1e-5f) * g[c] + be[c];
  if (lrelu) y = y > 0.f ? y : 0.2f * y;
  x[i] = y;
}

// ---------------------------------------------------------------------------
// softmax pieces + global top-128 over sal (raw-value top-k == softmax top-k)
// ---------------------------------------------------------------------------
__global__ void reduce_max_kernel(const float* __restrict__ v, long long n, float* out) {
  int tid = threadIdx.x; float m = -1e30f;
  for (long long i = tid; i < n; i += blockDim.x) m = fmaxf(m, v[i]);
  __shared__ float r[1024];
  r[tid] = m; __syncthreads();
  for (int o = 512; o > 0; o >>= 1) { if (tid < o) r[tid] = fmaxf(r[tid], r[tid + o]); __syncthreads(); }
  if (tid == 0) out[0] = r[0];
}

__global__ void reduce_sumexp_kernel(const float* __restrict__ v, long long n,
                                     const float* __restrict__ mx, float* out) {
  int tid = threadIdx.x; float s = 0.f; float m = mx[0];
  for (long long i = tid; i < n; i += blockDim.x) s += expf(v[i] - m);
  __shared__ float r[1024];
  r[tid] = s; __syncthreads();
  for (int o = 512; o > 0; o >>= 1) { if (tid < o) r[tid] += r[tid + o]; __syncthreads(); }
  if (tid == 0) out[0] = r[0];
}

__global__ void topk_kernel(float* __restrict__ v, long long n, int K,
                            int* __restrict__ idx_out, float* __restrict__ val_out) {
  int tid = threadIdx.x;
  __shared__ float rv[1024];
  __shared__ long long ri[1024];
  for (int k = 0; k < K; ++k) {
    float best = -1e30f; long long bi = 0;
    for (long long i = tid; i < n; i += blockDim.x) {
      float x = v[i];
      if (x > best) { best = x; bi = i; }
    }
    rv[tid] = best; ri[tid] = bi; __syncthreads();
    for (int o = 512; o > 0; o >>= 1) {
      if (tid < o && rv[tid + o] > rv[tid]) { rv[tid] = rv[tid + o]; ri[tid] = ri[tid + o]; }
      __syncthreads();
    }
    if (tid == 0) { idx_out[k] = (int)ri[0]; val_out[k] = rv[0]; v[ri[0]] = -1e30f; }
    __syncthreads();
  }
}

// s-value at the 128 surviving positions: softmax(sal) * sig, sig computed lazily.
__global__ void sparse_entries_kernel(const int* __restrict__ idx, const float* __restrict__ raw,
                                      const float* __restrict__ mx, const float* __restrict__ sum,
                                      const float* __restrict__ hF, const float* __restrict__ upw,
                                      const float* __restrict__ upb,
                                      int* cOut, int* tOut, float* sOut) {
  int j = threadIdx.x;                      // 0..127
  int fi = idx[j];
  int c = fi / N_SAMP, t = fi - c * N_SAMP;
  float soft = expf(raw[j] - mx[0]) / sum[0];
  float sig = upb[c];
  for (int i = 0; i < CH; ++i) sig += upw[c * CH + i] * hF[(size_t)i * N_SAMP + t];
  cOut[j] = c; tOut[j] = t; sOut[j] = soft * sig;
}

// agg = per-channel sum of s; g = agg @ W^T + b; lat = latents + g
__global__ void agg_lat_kernel(const int* __restrict__ cIdx, const float* __restrict__ sVal,
                               const float* __restrict__ ew, const float* __restrict__ eb,
                               const float* __restrict__ latents, float* __restrict__ lat) {
  __shared__ float agg[ENC];
  __shared__ float g[LATENT];
  int tid = threadIdx.x;                    // 512 threads
  agg[tid] = 0.f; __syncthreads();
  if (tid < KSP) atomicAdd(&agg[cIdx[tid]], sVal[tid]);
  __syncthreads();
  if (tid < LATENT) {
    float acc = eb[tid];
    for (int c = 0; c < ENC; ++c) acc += ew[tid * ENC + c] * agg[c];
    g[tid] = acc;
  }
  __syncthreads();
  for (int i = tid; i < ENC * LATENT; i += blockDim.x) lat[i] = latents[i] + g[i & (LATENT - 1)];
}

// lat[512,128] @ lin_w^T -> [512,512] == [512 atoms][128 ch][4 t]
__global__ void lin_kernel(const float* __restrict__ lat, const float* __restrict__ w,
                           const float* __restrict__ b, float* __restrict__ out) {
  int i = blockIdx.x * blockDim.x + threadIdx.x;
  if (i >= ENC * 512) return;
  int a = i >> 9, j = i & 511;
  float acc = b[j];
  const float* lr = lat + a * LATENT;
  const float* wr = w + (size_t)j * LATENT;
  for (int k = 0; k < LATENT; ++k) acc += lr[k] * wr[k];
  out[(size_t)a * 512 + j] = acc;
}

// to_dict final conv: 128 -> 1 channel, taps=3 pad 1, per atom, L=1024
__global__ void final1_conv_kernel(const float* __restrict__ in, const float* __restrict__ w,
                                   const float* __restrict__ b, float* __restrict__ dict) {
  int a = blockIdx.y;
  int t = blockIdx.x * blockDim.x + threadIdx.x;
  if (t >= KERNEL_L) return;
  const float* ap = in + (size_t)a * CH * KERNEL_L;
  float acc = b[0];
  for (int c = 0; c < CH; ++c) {
    const float* row = ap + (size_t)c * KERNEL_L;
    float xm = (t > 0) ? row[t - 1] : 0.f;
    float x0 = row[t];
    float xp = (t < KERNEL_L - 1) ? row[t + 1] : 0.f;
    acc += w[c * 3 + 0] * xm + w[c * 3 + 1] * x0 + w[c * 3 + 2] * xp;
  }
  dict[(size_t)a * KERNEL_L + t] = acc;
}

__global__ void unitnorm_kernel(float* __restrict__ dict) {
  int a = blockIdx.x, tid = threadIdx.x;
  float* p = dict + (size_t)a * KERNEL_L;
  float s = 0.f;
  for (int i = tid; i < KERNEL_L; i += 256) { float v = p[i]; s += v * v; }
  __shared__ float r[256];
  r[tid] = s; __syncthreads();
  for (int o = 128; o > 0; o >>= 1) { if (tid < o) r[tid] += r[tid + o]; __syncthreads(); }
  float inv = 1.f / (sqrtf(r[0]) + 1e-8f);
  for (int i = tid; i < KERNEL_L; i += 256) p[i] *= inv;
}

// sparse (128 nonzeros) (x) 1024-tap atoms, summed over channels == the fft_convolve
__global__ void scatter_conv_kernel(const int* __restrict__ cIdx, const int* __restrict__ tIdx,
                                    const float* __restrict__ sVal, const float* __restrict__ dict,
                                    float* __restrict__ y) {
  int t = blockIdx.x * blockDim.x + threadIdx.x;
  if (t >= N_SAMP) return;
  float acc = 0.f;
  for (int j = 0; j < KSP; ++j) {
    int dt = t - tIdx[j];
    if (dt >= 0 && dt < KERNEL_L) acc += sVal[j] * dict[(size_t)cIdx[j] * KERNEL_L + dt];
  }
  y[t] = acc;
}

__global__ void norm_y_kernel(float* __restrict__ y, int n) {
  int tid = threadIdx.x;
  __shared__ float r[1024];
  float m = 0.f;
  for (int i = tid; i < n; i += 1024) m = fmaxf(m, fabsf(y[i]));
  r[tid] = m; __syncthreads();
  for (int o = 512; o > 0; o >>= 1) { if (tid < o) r[tid] = fmaxf(r[tid], r[tid + o]); __syncthreads(); }
  float inv = 1.f / (r[0] + 1e-8f);
  for (int i = tid; i < n; i += 1024) y[i] *= inv;
}

// ---------------------------------------------------------------------------
static void launch_conv(const float* in, const float* w, const float* b, float* out,
                        _Float16* wpack, int Cout, int Lout, int Lin, int taps, int dil,
                        int ups, int lrelu, int atoms, hipStream_t st) {
  int total = Cout * taps * 128;             // packed fragment element count
  pack_w_kernel<<<(total + 255) / 256, 256, 0, st>>>(w, wpack, Cout, taps);
  dim3 grid((Lout + 63) / 64, Cout / 16, atoms);
  conv_wmma_kernel<<<grid, 32, 0, st>>>(in, wpack, b, out, Cout, Lout, Lin, taps, dil, ups, lrelu);
}

extern "C" void kernel_launch(void* const* d_in, const int* in_sizes, int n_in,
                              void* d_out, int out_size, void* d_ws, size_t ws_size,
                              hipStream_t stream) {
  auto F = [&](int k) { return (const float*)d_in[k]; };
  const float* x  = F(0);
  const float* fb = F(1);
  // input flat order (setup_inputs insertion order, recursive):
  // 2,3: analysis.up w,b ; 4..45: 7 blocks x {c1w,c1b,c2w,c2b,g,be}
  // 46..77: 8 expand layers x {cw,cb,g,be} ; 78,79: expand.final w,b
  // 80..83: sparse up w,b, sal w,b ; 84,85: embed w,b ; 86: latents
  // 87,88: to_dict lin w,b ; 89..120: 8 layers x {cw,cb,g,be} ; 121,122: final w,b
  const int DIL[7] = {1, 3, 9, 1, 3, 9, 1};

  char* ws = (char*)d_ws;
  float* pooled = (float*)(ws + 0x00000);       // 128*128
  float* stats  = (float*)(ws + 0x10000);       // 2*128
  float* maxp   = (float*)(ws + 0x11000);
  float* sump   = (float*)(ws + 0x11040);
  int*   kIdx   = (int*)  (ws + 0x11100);       // 128
  float* kVal   = (float*)(ws + 0x11300);       // 128
  int*   cArr   = (int*)  (ws + 0x11500);
  int*   tArr   = (int*)  (ws + 0x11700);
  float* sArr   = (float*)(ws + 0x11900);
  float* lat    = (float*)(ws + 0x12000);       // 512*128
  float* dict   = (float*)(ws + 0x60000);       // 512*1024
  _Float16* wpk = (_Float16*)(ws + 0x260000);   // packed A fragments (<=128KB)
  char*  arena  = ws + 0x400000;                // 4 MB header
  float* bufA = (float*)(arena);                               // 16 MB
  float* bufB = (float*)(arena + 0x1000000);                   // 16 MB
  float* bufC = (float*)(arena + 0x2000000);                   // small
  float* sal  = (float*)(arena + 0x2100000);                   // 64 MB
  float* tdA  = (float*)(arena);                               // 268 MB (aliases bufA..sal; dead by then)
  float* tdB  = (float*)(arena + 0x10000000);                  // 268 MB

  // ---- filter bank + pool ----
  fb_pool_kernel<<<dim3(128, 128), 256, 0, stream>>>(x, fb, pooled);

  // ---- analysis: 1x1 up + 7 dilated residual blocks (BN batch stats) ----
  launch_conv(pooled, F(2), F(3), bufA, wpk, 128, 128, 128, 1, 1, 0, 0, 1, stream);
  for (int bidx = 0; bidx < 7; ++bidx) {
    int base = 4 + bidx * 6, d = DIL[bidx];
    launch_conv(bufA, F(base + 0), F(base + 1), bufB, wpk, 128, 128, 128, 3, d, 0, 0, 1, stream);
    launch_conv(bufB, F(base + 2), F(base + 3), bufC, wpk, 128, 128, 128, 1, 1, 0, 0, 1, stream);
    add_lrelu_kernel<<<(16384 + 255) / 256, 256, 0, stream>>>(bufC, bufA, bufA, 16384);
    bn_stats_kernel<<<128, 256, 0, stream>>>(bufA, stats, 128, 128, 1);
    bn_apply_kernel<<<(16384 + 255) / 256, 256, 0, stream>>>(bufA, stats, F(base + 4), F(base + 5),
                                                             128, 128, 16384, 0);
  }

  // ---- expand: 8 x (x2 upsample, conv3, BN, lrelu), then final conv3 ----
  float* cur = bufA; float* nxt = bufB; int L = 128;
  for (int l = 0; l < 8; ++l) {
    int base = 46 + l * 4;
    launch_conv(cur, F(base + 0), F(base + 1), nxt, wpk, 128, 2 * L, L, 3, 1, 1, 0, 1, stream);
    L *= 2;
    long long ne = 128LL * L;
    bn_stats_kernel<<<128, 256, 0, stream>>>(nxt, stats, 128, L, 1);
    bn_apply_kernel<<<(unsigned)((ne + 255) / 256), 256, 0, stream>>>(nxt, stats, F(base + 2),
                                                                      F(base + 3), 128, L, ne, 1);
    float* tmp = cur; cur = nxt; nxt = tmp;
  }
  launch_conv(cur, F(78), F(79), nxt, wpk, 128, L, L, 3, 1, 0, 0, 1, stream);   // L = 32768
  float* hF = nxt;

  // ---- saliency conv, softmax pieces, global top-128, sparse values ----
  launch_conv(hF, F(82), F(83), sal, wpk, 512, N_SAMP, N_SAMP, 1, 1, 0, 0, 1, stream);
  long long nsal = (long long)ENC * N_SAMP;
  reduce_max_kernel<<<1, 1024, 0, stream>>>(sal, nsal, maxp);
  reduce_sumexp_kernel<<<1, 1024, 0, stream>>>(sal, nsal, maxp, sump);
  topk_kernel<<<1, 1024, 0, stream>>>(sal, nsal, KSP, kIdx, kVal);
  sparse_entries_kernel<<<1, 128, 0, stream>>>(kIdx, kVal, maxp, sump, hF, F(80), F(81),
                                               cArr, tArr, sArr);
  agg_lat_kernel<<<1, 512, 0, stream>>>(cArr, sArr, F(84), F(85), F(86), lat);

  // ---- to_dict: linear -> [512,128,4], 8 x (x2 up, conv3, BN over atoms+time, lrelu) ----
  lin_kernel<<<(ENC * 512 + 255) / 256, 256, 0, stream>>>(lat, F(87), F(88), tdA);
  float* c2 = tdA; float* n2 = tdB; int Lt = 4;
  for (int l = 0; l < 8; ++l) {
    int base = 89 + l * 4;
    launch_conv(c2, F(base + 0), F(base + 1), n2, wpk, 128, 2 * Lt, Lt, 3, 1, 1, 0, ENC, stream);
    Lt *= 2;
    long long ne = (long long)ENC * 128 * Lt;
    bn_stats_kernel<<<128, 256, 0, stream>>>(n2, stats, 128, Lt, ENC);
    bn_apply_kernel<<<(unsigned)((ne + 255) / 256), 256, 0, stream>>>(n2, stats, F(base + 2),
                                                                      F(base + 3), 128, Lt, ne, 1);
    float* tmp = c2; c2 = n2; n2 = tmp;
  }
  final1_conv_kernel<<<dim3(KERNEL_L / 256, ENC), 256, 0, stream>>>(c2, F(121), F(122), dict);
  unitnorm_kernel<<<512, 256, 0, stream>>>(dict);

  // ---- sparse time-domain convolution (== fft_convolve) + max-norm ----
  float* y = (float*)d_out;
  scatter_conv_kernel<<<(N_SAMP + 255) / 256, 256, 0, stream>>>(cArr, tArr, sArr, dict, y);
  norm_y_kernel<<<1, 1024, 0, stream>>>(y, N_SAMP);
}
